// ChannelMHSA_5738076307559
// MI455X (gfx1250) — matrix operand order repaired and checked by reading, not compiled, not run on hardware
//
#include <hip/hip_runtime.h>

typedef __attribute__((ext_vector_type(4)))  _Float16 v4h;
typedef __attribute__((ext_vector_type(8)))  _Float16 v8h;
typedef __attribute__((ext_vector_type(16))) _Float16 v16h;
typedef __attribute__((ext_vector_type(8)))  float    v8f;
typedef __attribute__((ext_vector_type(4)))  unsigned v4u;
typedef __attribute__((ext_vector_type(8)))  int      v8i;
typedef __attribute__((ext_vector_type(4)))  int      v4i;

union U16H { v16h v; v8h g[2]; };

__device__ __forceinline__ unsigned lds_off(const void* p) {
  // low 32 bits of a generic pointer to LDS == DS byte offset
  return (unsigned)(size_t)p;
}
__device__ __forceinline__ void async_cp_b128(unsigned dst, const void* src) {
  asm volatile("global_load_async_to_lds_b128 %0, %1, off"
               :: "v"(dst), "v"(src) : "memory");
}
__device__ __forceinline__ void wait_async0() {
  asm volatile("s_wait_asynccnt 0x0" ::: "memory");
}

// ---------------------------------------------------------------------------
// TDM: load a 2D tile (tile_d0 contiguous f16 elems x tile_d1 rows) from a
// row-major tensor with row stride `stride0` (elems) into LDS, padding the
// LDS destination by 32B after every 64B row chunk (=> stride-48-half rows).
// D# bit packing per CDNA5 ISA.
// ---------------------------------------------------------------------------
__device__ __forceinline__ void tdm_load_2d(unsigned lds_byte_off, const void* gaddr,
                                            unsigned tensor_d0, unsigned tensor_d1,
                                            unsigned tile_d0, unsigned tile_d1,
                                            unsigned long stride0) {
  const unsigned long ga = (unsigned long)gaddr;
  v4u g0;
  g0.x = 0x1u;                                            // count=1, load, user
  g0.y = lds_byte_off;                                    // lds_addr
  g0.z = (unsigned)(ga & 0xffffffffu);                    // global_addr[31:0]
  g0.w = (unsigned)((ga >> 32) & 0x1ffffffu) | (2u << 30); // addr[56:32] | type=2
  v8i g1;
  // data_size=1 (2B), pad_enable, pad_interval=3 (16 DW = 64B), pad_amount=7 (8 DW = 32B)
  g1[0] = (int)((1u << 16) | (1u << 20) | (3u << 22) | (7u << 25));
  g1[1] = (int)((tensor_d0 & 0xffffu) << 16);             // [63:48]=tensor_dim0.lo
  g1[2] = (int)(((tensor_d0 >> 16) & 0xffffu) | ((tensor_d1 & 0xffffu) << 16));
  g1[3] = (int)(((tensor_d1 >> 16) & 0xffffu) | ((tile_d0 & 0xffffu) << 16));
  g1[4] = (int)(tile_d1 & 0xffffu);                       // tile_dim1, tile_dim2=0
  g1[5] = (int)(stride0 & 0xffffffffu);                   // tensor_dim0_stride.lo
  g1[6] = (int)((stride0 >> 32) & 0xffffu);               // stride0.hi, stride1=0
  g1[7] = 0;
  const v4i z4 = {0, 0, 0, 0};
  const v8i z8 = {0, 0, 0, 0, 0, 0, 0, 0};
  __builtin_amdgcn_tensor_load_to_lds(g0, g1, z4, z4, z8, 0);
}

// ---------------------------------------------------------------------------
// Prep: f32 -> f16 elementwise (for x)
// ---------------------------------------------------------------------------
__global__ __launch_bounds__(256) void cvt_f16(const float* __restrict__ in,
                                               _Float16* __restrict__ out, int n) {
  int i = (blockIdx.x * 256 + threadIdx.x) * 4;
  if (i + 3 < n) {
    float4 t = *(const float4*)&in[i];
    v4h s = { (_Float16)t.x, (_Float16)t.y, (_Float16)t.z, (_Float16)t.w };
    *(v4h*)&out[i] = s;
  }
}

// ---------------------------------------------------------------------------
// Prep: transpose+convert weights: in (K x N f32) -> out (N x K f16)
// ---------------------------------------------------------------------------
__global__ __launch_bounds__(256) void transpose_cvt(const float* __restrict__ in,
                                                     _Float16* __restrict__ out,
                                                     int K, int N) {
  __shared__ float tile[32][33];
  const int k0 = blockIdx.y * 32, n0 = blockIdx.x * 32;
  const int tx = threadIdx.x & 31, ty = threadIdx.x >> 5; // 32 x 8
  #pragma unroll
  for (int r = 0; r < 32; r += 8)
    tile[ty + r][tx] = in[(size_t)(k0 + ty + r) * N + n0 + tx];
  __syncthreads();
  #pragma unroll
  for (int r = 0; r < 32; r += 8)
    out[(size_t)(n0 + ty + r) * K + k0 + tx] = (_Float16)tile[tx][ty + r];
}

// ---------------------------------------------------------------------------
// Tiled GEMM: C(MxN) = A(MxK) @ Bt(NxK)^T + bias(N), all-f16 operands.
// Block 256 threads (8 waves, 4x2), tile 128x128, K-step 32, double-buffered.
// B tile: one TDM tensor_load_to_lds (wave 0) with LDS padding.
// A tile: per-lane global_load_async_to_lds_b128.
// Both overlap the WMMA work of the current tile; synced via
// s_wait_asynccnt / s_wait_tensorcnt before the barrier.
// ---------------------------------------------------------------------------
template <typename OT>
__global__ __launch_bounds__(256) void gemm_wmma_f16(
    const _Float16* __restrict__ A,   // M x K row-major f16
    const _Float16* __restrict__ Bt,  // N x K row-major f16 (B transposed)
    const float* __restrict__ bias, OT* __restrict__ C,
    int M, int N, int K)
{
  __shared__ _Float16 Alds[2][128 * 48]; // [m][k], row stride 48 halfs (96B)
  __shared__ _Float16 Blds[2][128 * 48]; // [n][k]

  const int tid  = threadIdx.x;
  const int lane = tid & 31, wave = tid >> 5;
  const int lrow = lane & 15, lhi = lane >> 4;
  const int wm = wave >> 1, wn = wave & 1;
  const int m0 = blockIdx.y * 128, n0 = blockIdx.x * 128;

  v8f acc[2][4] = {};

  auto stage = [&](int buf, int kt) {
    if (wave == 0)  // whole 128x32 B tile in one TDM op
      tdm_load_2d(lds_off(&Blds[buf][0]), &Bt[(size_t)n0 * K + kt],
                  (unsigned)(K - kt), (unsigned)(N - n0), 32u, 128u,
                  (unsigned long)K);
    #pragma unroll
    for (int t = 0; t < 2; ++t) {  // A tile: 512 chunks of 8 halfs
      const int c = tid + t * 256;
      const int row = c >> 2, k8 = (c & 3) << 3;
      async_cp_b128(lds_off(&Alds[buf][row * 48 + k8]),
                    &A[(size_t)(m0 + row) * K + kt + k8]);
    }
  };

  stage(0, 0);
  wait_async0();
  if (wave == 0) __builtin_amdgcn_s_wait_tensorcnt(0);
  __syncthreads();

  int buf = 0;
  for (int kt = 0; kt < K; kt += 32) {
    if (kt + 32 < K) stage(buf ^ 1, kt + 32);   // overlap next copy with compute

    v16h af[2], bf[4];
    #pragma unroll
    for (int i = 0; i < 2; ++i) {
      const int r = (wm * 32 + i * 16 + lrow) * 48;
      U16H u;
      u.g[0] = *(const v8h*)&Alds[buf][r + lhi * 8];
      u.g[1] = *(const v8h*)&Alds[buf][r + 16 + lhi * 8];
      af[i] = u.v;
    }
    #pragma unroll
    for (int j = 0; j < 4; ++j) {
      const int r = (wn * 64 + j * 16 + lrow) * 48;
      U16H u;
      u.g[0] = *(const v8h*)&Blds[buf][r + lhi * 8];
      u.g[1] = *(const v8h*)&Blds[buf][r + 16 + lhi * 8];
      bf[j] = u.v;
    }
    #pragma unroll
    for (int i = 0; i < 2; ++i)
      #pragma unroll
      for (int j = 0; j < 4; ++j)
        acc[i][j] = __builtin_amdgcn_wmma_f32_16x16x32_f16(
            false, af[i], false, bf[j], (short)0, acc[i][j], false, false);

    wait_async0();
    if (wave == 0) __builtin_amdgcn_s_wait_tensorcnt(0);
    __syncthreads();
    buf ^= 1;
  }

  #pragma unroll
  for (int i = 0; i < 2; ++i) {
    #pragma unroll
    for (int j = 0; j < 4; ++j) {
      const int gm = m0 + wm * 32 + i * 16 + lhi * 8;
      const int gn = n0 + wn * 64 + j * 16 + lrow;
      const float bv = bias[gn];
      #pragma unroll
      for (int r = 0; r < 8; ++r) {
        if constexpr (__is_same(OT, float))
          C[(size_t)(gm + r) * N + gn] = acc[i][j][r] + bv;
        else
          C[(size_t)(gm + r) * N + gn] = (_Float16)(acc[i][j][r] + bv);
      }
    }
  }
}

// ---------------------------------------------------------------------------
// Channel attention per (b,h): S = (k^T v)/8, softmax rows, out1 = q @ attn^T
// qkv layout: [b*1024+n][2304] f16; q at col h*64, k at 768+h*64, v at 1536+h*64
// ---------------------------------------------------------------------------
__global__ __launch_bounds__(256) void chan_attn_wmma(
    const _Float16* __restrict__ qkv, _Float16* __restrict__ out1)
{
  __shared__ _Float16 kT[64 * 48];   // [d][n-chunk]
  __shared__ _Float16 vT[64 * 48];   // [e][n-chunk]
  __shared__ float    S[64 * 64];
  __shared__ _Float16 attn[64 * 72]; // softmaxed, row-major [d][e]

  const int bh = blockIdx.x;
  const int b = bh / 12, h = bh % 12;
  const size_t base = (size_t)b * 1024 * 2304;
  const int qoff = h * 64, koff = 768 + h * 64, voff = 1536 + h * 64;

  const int tid = threadIdx.x;
  const int lane = tid & 31, wave = tid >> 5;
  const int lrow = lane & 15, lhi = lane >> 4;

  // ---- phase 1: S = (k^T v) * 0.125 ; waves tile (d,e) as 4x2 ----
  const int wd = wave >> 1, we = wave & 1;
  v8f sacc[2] = {};
  for (int nt = 0; nt < 1024; nt += 32) {
    #pragma unroll
    for (int t = 0; t < 2; ++t) {
      const int c = tid + t * 256;  // 0..511
      const int n = c >> 4, d4 = (c & 15) << 2;
      v4h kv = *(const v4h*)&qkv[base + (size_t)(nt + n) * 2304 + koff + d4];
      v4h vv = *(const v4h*)&qkv[base + (size_t)(nt + n) * 2304 + voff + d4];
      kT[(d4 + 0) * 48 + n] = kv.x; kT[(d4 + 1) * 48 + n] = kv.y;
      kT[(d4 + 2) * 48 + n] = kv.z; kT[(d4 + 3) * 48 + n] = kv.w;
      vT[(d4 + 0) * 48 + n] = vv.x; vT[(d4 + 1) * 48 + n] = vv.y;
      vT[(d4 + 2) * 48 + n] = vv.z; vT[(d4 + 3) * 48 + n] = vv.w;
    }
    __syncthreads();
    U16H ua;
    const int ar = (wd * 16 + lrow) * 48;
    ua.g[0] = *(const v8h*)&kT[ar + lhi * 8];
    ua.g[1] = *(const v8h*)&kT[ar + 16 + lhi * 8];
    #pragma unroll
    for (int j = 0; j < 2; ++j) {
      const int br = (we * 32 + j * 16 + lrow) * 48;
      U16H ub;
      ub.g[0] = *(const v8h*)&vT[br + lhi * 8];
      ub.g[1] = *(const v8h*)&vT[br + 16 + lhi * 8];
      sacc[j] = __builtin_amdgcn_wmma_f32_16x16x32_f16(
          false, ua.v, false, ub.v, (short)0, sacc[j], false, false);
    }
    __syncthreads();
  }
  #pragma unroll
  for (int j = 0; j < 2; ++j) {
    const int d = wd * 16 + lhi * 8;
    const int e = we * 32 + j * 16 + lrow;
    #pragma unroll
    for (int r = 0; r < 8; ++r) S[(d + r) * 64 + e] = sacc[j][r] * 0.125f;
  }
  __syncthreads();

  // ---- softmax over e per row d (one thread per row) ----
  if (tid < 64) {
    float mx = -1e30f;
    for (int e = 0; e < 64; ++e) mx = fmaxf(mx, S[tid * 64 + e]);
    float sum = 0.f;
    for (int e = 0; e < 64; ++e) {
      float v = __expf(S[tid * 64 + e] - mx);
      S[tid * 64 + e] = v;
      sum += v;
    }
    float inv = 1.f / sum;
    for (int e = 0; e < 64; ++e)
      attn[tid * 72 + e] = (_Float16)(S[tid * 64 + e] * inv);
  }
  __syncthreads();

  // ---- phase 2: out1[n,d] = sum_e attn[d,e] q[n,e]  (= q @ attn^T) ----
  const int mbase = wave * 128;
  for (int ms = 0; ms < 8; ++ms) {
    const int mrow = mbase + ms * 16;
    const _Float16* qrow = &qkv[base + (size_t)(mrow + lrow) * 2304 + qoff];
    v8f oacc[4] = {};
    #pragma unroll
    for (int ktile = 0; ktile < 2; ++ktile) {
      U16H ua;  // A = q fragment straight from global (16B-aligned)
      ua.g[0] = *(const v8h*)&qrow[ktile * 32 + lhi * 8];
      ua.g[1] = *(const v8h*)&qrow[ktile * 32 + 16 + lhi * 8];
      #pragma unroll
      for (int j = 0; j < 4; ++j) {
        const int dcol = j * 16 + lrow;
        U16H ub; // B[e,d] = attn[d,e] -> row-major attn is column-contiguous
        ub.g[0] = *(const v8h*)&attn[dcol * 72 + ktile * 32 + lhi * 8];
        ub.g[1] = *(const v8h*)&attn[dcol * 72 + ktile * 32 + 16 + lhi * 8];
        oacc[j] = __builtin_amdgcn_wmma_f32_16x16x32_f16(
            false, ua.v, false, ub.v, (short)0, oacc[j], false, false);
      }
    }
    #pragma unroll
    for (int j = 0; j < 4; ++j) {
      const int e = j * 16 + lrow;
      const int mr = mrow + lhi * 8;
      #pragma unroll
      for (int r = 0; r < 8; ++r)
        out1[(size_t)(b * 1024 + mr + r) * 768 + h * 64 + e] = (_Float16)oacc[j][r];
    }
  }
}

// ---------------------------------------------------------------------------
extern "C" void kernel_launch(void* const* d_in, const int* in_sizes, int n_in,
                              void* d_out, int out_size, void* d_ws, size_t ws_size,
                              hipStream_t stream) {
  (void)in_sizes; (void)n_in; (void)out_size; (void)ws_size;
  const float* x     = (const float*)d_in[0];  // (32,1024,768)
  const float* w_qkv = (const float*)d_in[1];  // (768,2304)
  const float* b_qkv = (const float*)d_in[2];  // (2304,)
  const float* w_out = (const float*)d_in[3];  // (768,768)
  const float* b_out = (const float*)d_in[4];  // (768,)
  float* out = (float*)d_out;                  // (32,1024,768)

  const int M = 32 * 1024, C = 768, C3 = 2304;

  _Float16* xh     = (_Float16*)d_ws;                  // M x 768
  _Float16* wqkvT  = xh + (size_t)M * C;               // 2304 x 768
  _Float16* woutT  = wqkvT + (size_t)C3 * C;           // 768 x 768
  _Float16* qkvh   = woutT + (size_t)C * C;            // M x 2304
  _Float16* out1h  = qkvh + (size_t)M * C3;            // M x 768

  // prep: convert x, transpose+convert weights (one-time, ~110MB traffic)
  cvt_f16<<<dim3((M * C) / 1024), 256, 0, stream>>>(x, xh, M * C);
  transpose_cvt<<<dim3(C3 / 32, C / 32), 256, 0, stream>>>(w_qkv, wqkvT, C, C3);
  transpose_cvt<<<dim3(C / 32, C / 32), 256, 0, stream>>>(w_out, woutT, C, C);

  // 1) qkv = x @ w_qkv + b_qkv  (f16 out)
  gemm_wmma_f16<_Float16>
      <<<dim3(C3 / 128, M / 128), 256, 0, stream>>>(xh, wqkvT, b_qkv, qkvh, M, C3, C);

  // 2) per-(b,h) channel attention
  chan_attn_wmma<<<dim3(32 * 12), 256, 0, stream>>>(qkvh, out1h);

  // 3) out = out1 @ w_out + b_out  (f32 out)
  gemm_wmma_f16<float>
      <<<dim3(C / 128, M / 128), 256, 0, stream>>>(out1h, woutT, b_out, out, M, C, C);
}
